// AttentionLayer_74629351735962
// MI455X (gfx1250) — matrix operand order repaired and checked by reading, not compiled, not run on hardware
//
#include <hip/hip_runtime.h>
#include <stdint.h>

// Problem sizes (fixed by the reference)
#define B_  8
#define S_  16
#define P_  256
#define Q_  96
#define D_  768
#define G_  (4 * D_)
#define VNEG (-1e30f)

// Padded LDS row strides (in 16-bit elements) -> conflict-free ds_load_b128
#define UB  776    // u rows: 768 + 8
#define CB  104    // c2q rows: 96 + 8

// LDS byte offsets (dynamic LDS starts at offset 0; no static __shared__ here)
#define LDS_FP32_OFF   ((Q_ * UB + P_ * CB) * 2)                          // 202240
#define LDS_STAGE0_OFF (LDS_FP32_OFF + (3 * D_ + Q_ + 3 * P_ + D_ + 16) * 4) // 218048
#define LDS_STAGE1_OFF (LDS_STAGE0_OFF + 16 * D_ * 4)                     // 267200
#define LDS_TOTAL      (LDS_STAGE1_OFF + 16 * D_ * 4)                     // 316352

typedef __attribute__((ext_vector_type(16))) __bf16        v16bf;
typedef __attribute__((ext_vector_type(8)))  float         v8f;
typedef __attribute__((ext_vector_type(8)))  unsigned int  v8u;
typedef __attribute__((ext_vector_type(4)))  int           si4;
typedef __attribute__((ext_vector_type(8)))  int           si8;

// f32 -> bf16 (RTNE) through the native cast: lets the backend pick
// v_cvt_pk_bf16_f32-class instructions instead of a 3-op integer sequence.
__device__ __forceinline__ unsigned short f2bf(float x) {
    __bf16 b = (__bf16)x;
    return __builtin_bit_cast(unsigned short, b);
}
__device__ __forceinline__ unsigned packbf(float lo, float hi) {
    return (unsigned)f2bf(lo) | ((unsigned)f2bf(hi) << 16);
}
__device__ __forceinline__ float bf2f(unsigned short h) {
    unsigned int u = ((unsigned int)h) << 16;
    return __builtin_bit_cast(float, u);
}

// ---- Tensor Data Mover: DMA a 16x768 fp32 tile (one u chunk) into LDS ----
// D# layout per CDNA5 ISA (2 SGPR groups, 2D).
__device__ __forceinline__ void tdm_load_u_chunk(const float* gsrc, unsigned ldsOff) {
    unsigned long long ga = (unsigned long long)(const void*)gsrc;
    si4 g0;
    g0[0] = 1;                                   // count=1, user desc, no gather
    g0[1] = (int)ldsOff;                         // lds_addr (bytes)
    g0[2] = (int)(unsigned)(ga & 0xFFFFFFFFull); // global_addr[31:0]
    g0[3] = (int)((unsigned)((ga >> 32) & 0x01FFFFFFu) | 0x80000000u); // addr[56:32] | type=2
    si8 g1;
    g1[0] = 0x20000;                             // wg_mask=0 (not in cluster), data_size=4B
    g1[1] = (int)((D_ & 0xFFFF) << 16);          // tensor_dim0 lo16 @ bits[63:48]
    g1[2] = (int)((D_ >> 16) | (Q_ << 16));      // tensor_dim0 hi16, tensor_dim1 lo16
    g1[3] = (int)((Q_ >> 16) | (D_ << 16));      // tensor_dim1 hi16, tile_dim0=768
    g1[4] = 16;                                  // tile_dim1=16, tile_dim2=0
    g1[5] = D_;                                  // tensor_dim0_stride lo32
    g1[6] = (int)((D_ & 0xFFFF) << 16);          // stride0 hi16=0, stride1 lo16
    g1[7] = 0;                                   // stride1 hi32
    asm volatile("tensor_load_to_lds %0, %1" :: "s"(g0), "s"(g1) : "memory");
}
__device__ __forceinline__ void wait_tensor_le1(void) {
#if __has_builtin(__builtin_amdgcn_s_wait_tensorcnt)
    __builtin_amdgcn_s_wait_tensorcnt(1);
#else
    asm volatile("s_wait_tensorcnt 0x1" ::: "memory");
#endif
}
__device__ __forceinline__ void wait_tensor_le0(void) {
#if __has_builtin(__builtin_amdgcn_s_wait_tensorcnt)
    __builtin_amdgcn_s_wait_tensorcnt(0);
#else
    asm volatile("s_wait_tensorcnt 0x0" ::: "memory");
#endif
}

// One workgroup (256 threads = 8 wave32) per (b,s) tile.
// Wave w owns output rows p in [32w, 32w+32)  (two 16-row M tiles).
__global__ __launch_bounds__(256, 1)
void attn_bidaf_gfx1250(const float* __restrict__ h,  const float* __restrict__ u,
                        const float* __restrict__ hm, const float* __restrict__ um,
                        const float* __restrict__ w,  const float* __restrict__ bsc,
                        float* __restrict__ g)
{
    extern __shared__ __align__(16) char smem[];
    unsigned short* u_sh   = (unsigned short*)smem;           // Q_*UB bf16 (u, row-major)
    unsigned short* c2q_sh = u_sh + Q_ * UB;                  // P_*CB bf16 (softmax probs)
    float* w_sh   = (float*)(c2q_sh + P_ * CB);               // 3*D  (w_h | w_u | w_hu)
    float* uw_sh  = w_sh   + 3 * D_;                          // Q_
    float* hw_sh  = uw_sh  + Q_;                              // P_
    float* rmax_sh= hw_sh  + P_;                              // P_   (row max over q)
    float* q2c_sh = rmax_sh+ P_;                              // P_
    float* ha_sh  = q2c_sh + P_;                              // D_   (h_a)
    float* red_sh = ha_sh  + D_;                              // 16
    float* stage0 = red_sh + 16;                              // 16*D fp32 TDM staging A
    float* stage1 = stage0 + 16 * D_;                         // 16*D fp32 TDM staging B

    const int tid  = threadIdx.x;
    const int lane = tid & 31;
    const int wid  = tid >> 5;
    const int ln15 = lane & 15;
    const int hi   = lane >> 4;          // which 16-lane half

    const int bs = blockIdx.x;
    const int b  = bs / S_;
    const float* hB  = h  + (size_t)bs * P_ * D_;
    const float* uB  = u  + (size_t)b  * Q_ * D_;
    const float* hmB = hm + (size_t)bs * P_;
    const float* umB = um + (size_t)b  * Q_;
    float*       gB  = g  + (size_t)bs * P_ * G_;
    const float bias = bsc[0];

    // ---------------- Phase 1: TDM-stage u (fp32) -> convert to bf16 --------
    // Kick the first DMA immediately so it overlaps the w/ha setup below.
    if (wid == 0) tdm_load_u_chunk(uB, LDS_STAGE0_OFF);

    for (int i = tid; i < 3 * D_; i += 256) w_sh[i] = w[i];
    for (int i = tid; i < D_; i += 256) ha_sh[i] = 0.f;

    for (int c = 0; c < 6; ++c) {                 // 6 chunks of 16 u-rows
        const float* buf = (c & 1) ? stage1 : stage0;
        if (wid == 0) {
            if (c + 1 < 6) {                      // prefetch next chunk (double buffer)
                tdm_load_u_chunk(uB + (size_t)(c + 1) * 16 * D_,
                                 ((c + 1) & 1) ? LDS_STAGE1_OFF : LDS_STAGE0_OFF);
                wait_tensor_le1();                // chunk c complete, c+1 in flight
            } else {
                wait_tensor_le0();
            }
        }
        __syncthreads();                          // chunk c visible to all waves
        for (int r = 0; r < 16; ++r) {
            const int q = c * 16 + r;
            for (int d = tid; d < D_; d += 256)
                u_sh[q * UB + d] = f2bf(buf[r * D_ + d]);
        }
        __syncthreads();                          // done with buf before it is reused
    }

    // uw[q] = u[q,:] . w_u   (threads 0..95, one row each)
    if (tid < Q_) {
        float acc = 0.f;
        for (int d = 0; d < D_; ++d) acc += bf2f(u_sh[tid * UB + d]) * w_sh[D_ + d];
        uw_sh[tid] = acc;
    }

    // ---------------- Phase 2: GEMM1  M = (h*w_hu) @ u^T  (256x96, K=768) ----
    const int mbase = wid * 32;
    const v8f z8 = {0.f, 0.f, 0.f, 0.f, 0.f, 0.f, 0.f, 0.f};
    v8f acc1[2][6];
#pragma unroll
    for (int mt = 0; mt < 2; ++mt)
#pragma unroll
        for (int nt = 0; nt < 6; ++nt) acc1[mt][nt] = z8;

    const int k0 = hi * 8;   // A-layout: lane half selects K sub-offset
    const float* arow0 = hB + (size_t)(mbase + ln15) * D_;
    const float* arow1 = hB + (size_t)(mbase + 16 + ln15) * D_;
    float hwp0 = 0.f, hwp1 = 0.f;   // partial h.w_h per lane

    for (int ks = 0; ks < 24; ++ks) {
        const int kb = ks * 32;
        __builtin_prefetch(arow0 + kb + 64, 0, 0);   // global_prefetch_b8
        __builtin_prefetch(arow1 + kb + 64, 0, 0);
        v16bf afr[2];
#pragma unroll
        for (int mt = 0; mt < 2; ++mt) {
            const float* rp = (mt ? arow1 : arow0) + kb + k0;
            float4 f0 = *(const float4*)(rp);
            float4 f1 = *(const float4*)(rp + 4);
            float4 f2 = *(const float4*)(rp + 16);
            float4 f3 = *(const float4*)(rp + 20);
            float fv[16] = {f0.x, f0.y, f0.z, f0.w, f1.x, f1.y, f1.z, f1.w,
                            f2.x, f2.y, f2.z, f2.w, f3.x, f3.y, f3.z, f3.w};
            float hp = 0.f;
#pragma unroll
            for (int i = 0; i < 16; ++i) {
                const int kk = kb + k0 + i + ((i >= 8) ? 8 : 0);
                hp += fv[i] * w_sh[kk];          // h . w_h partial
                fv[i] *= w_sh[2 * D_ + kk];      // scale by w_hu
            }
            v8u av;
#pragma unroll
            for (int j = 0; j < 8; ++j)
                av[j] = packbf(fv[2 * j], fv[2 * j + 1]);
            afr[mt] = __builtin_bit_cast(v16bf, av);
            if (mt == 0) hwp0 += hp; else hwp1 += hp;
        }
#pragma unroll
        for (int nt = 0; nt < 6; ++nt) {
            // B fragment: column n = u row (nt*16+ln15); K = 8 at hi*8, 8 at 16+hi*8
            const uint4* bp = (const uint4*)&u_sh[(nt * 16 + ln15) * UB + kb + hi * 8];
            uint4 b0 = bp[0], b1 = bp[2];
            v8u bv; bv[0]=b0.x; bv[1]=b0.y; bv[2]=b0.z; bv[3]=b0.w;
                    bv[4]=b1.x; bv[5]=b1.y; bv[6]=b1.z; bv[7]=b1.w;
            v16bf bfr = __builtin_bit_cast(v16bf, bv);
            acc1[0][nt] = __builtin_amdgcn_wmma_f32_16x16x32_bf16(
                false, afr[0], false, bfr, (short)0, acc1[0][nt], false, false);
            acc1[1][nt] = __builtin_amdgcn_wmma_f32_16x16x32_bf16(
                false, afr[1], false, bfr, (short)0, acc1[1][nt], false, false);
        }
    }
    // combine the two K-halves of the h.w_h dot product
    float hw0 = hwp0 + __shfl_xor(hwp0, 16, 32);
    float hw1 = hwp1 + __shfl_xor(hwp1, 16, 32);
    if (hi == 0) { hw_sh[mbase + ln15] = hw0; hw_sh[mbase + 16 + ln15] = hw1; }
    __syncthreads();

    // ---------------- Phase 3: logits epilogue + softmax over q -------------
    float umv[6];
#pragma unroll
    for (int nt = 0; nt < 6; ++nt) umv[nt] = umB[nt * 16 + ln15];
#pragma unroll
    for (int mt = 0; mt < 2; ++mt) {
#pragma unroll
        for (int r = 0; r < 8; ++r) {
            // C layout: lane half hi holds rows r (hi=0) / r+8 (hi=1), n = ln15
            const int m = mbase + mt * 16 + r + 8 * hi;
            const float hwv = hw_sh[m];
            const float hmv = hmB[m];
            float lg[6]; float mx = -3.4e38f;
#pragma unroll
            for (int nt = 0; nt < 6; ++nt) {
                float v = acc1[mt][nt][r] + hwv + uw_sh[nt * 16 + ln15] + bias
                        + (1.0f - hmv * umv[nt]) * VNEG;
                lg[nt] = v; mx = fmaxf(mx, v);
            }
#pragma unroll
            for (int o = 8; o >= 1; o >>= 1) mx = fmaxf(mx, __shfl_xor(mx, o, 32));
            if (ln15 == 0) rmax_sh[m] = mx;
            float sm = 0.f;
#pragma unroll
            for (int nt = 0; nt < 6; ++nt) { lg[nt] = __expf(lg[nt] - mx); sm += lg[nt]; }
#pragma unroll
            for (int o = 8; o >= 1; o >>= 1) sm += __shfl_xor(sm, o, 32);
            const float inv = 1.0f / sm;
#pragma unroll
            for (int nt = 0; nt < 6; ++nt)
                c2q_sh[m * CB + nt * 16 + ln15] = f2bf(lg[nt] * inv);
        }
    }
    __syncthreads();

    // ---------------- Phase 4: q2c = softmax_p(rowmax) ----------------------
    {
        float v = rmax_sh[tid];
        float m = v;
#pragma unroll
        for (int o = 16; o >= 1; o >>= 1) m = fmaxf(m, __shfl_xor(m, o, 32));
        if (lane == 0) red_sh[wid] = m;
        __syncthreads();
        float gm = red_sh[0];
        for (int i = 1; i < 8; ++i) gm = fmaxf(gm, red_sh[i]);
        float e = __expf(v - gm);
        float sm = e;
#pragma unroll
        for (int o = 16; o >= 1; o >>= 1) sm += __shfl_xor(sm, o, 32);
        if (lane == 0) red_sh[8 + wid] = sm;
        __syncthreads();
        float tot = 0.f;
        for (int i = 0; i < 8; ++i) tot += red_sh[8 + i];
        q2c_sh[tid] = e / tot;
    }
    __syncthreads();

    // ---------------- Phase 5: h_a[d] = sum_p q2c[p] * h[p,d] ---------------
    {
        float ha[24];
#pragma unroll
        for (int j = 0; j < 24; ++j) ha[j] = 0.f;
        for (int pr = 0; pr < 32; ++pr) {
            const int p = mbase + pr;
            const float qv = q2c_sh[p];
            const float* rp = hB + (size_t)p * D_;
#pragma unroll
            for (int j = 0; j < 24; ++j) ha[j] += qv * rp[lane + 32 * j];
        }
#pragma unroll
        for (int j = 0; j < 24; ++j) atomicAdd(&ha_sh[lane + 32 * j], ha[j]);
    }
    __syncthreads();

    // ---------------- Phase 6: GEMM2 u_a = c2q @ u, write g -----------------
    // A fragments (c2q rows) are invariant over the 48 d-tiles: preload.
    v16bf a2[3][2];
#pragma unroll
    for (int ksq = 0; ksq < 3; ++ksq) {
#pragma unroll
        for (int mt = 0; mt < 2; ++mt) {
            const int p  = mbase + mt * 16 + ln15;
            const int qb = ksq * 32 + hi * 8;
            const uint4* ap0 = (const uint4*)&c2q_sh[p * CB + qb];
            const uint4* ap1 = (const uint4*)&c2q_sh[p * CB + qb + 16];
            uint4 x0 = ap0[0], x1 = ap1[0];
            v8u av; av[0]=x0.x; av[1]=x0.y; av[2]=x0.z; av[3]=x0.w;
                    av[4]=x1.x; av[5]=x1.y; av[6]=x1.z; av[7]=x1.w;
            a2[ksq][mt] = __builtin_bit_cast(v16bf, av);
        }
    }

    for (int nt2 = 0; nt2 < 48; ++nt2) {
        const int d0   = nt2 * 16;
        const int dcol = d0 + ln15;
        v8f acc2[2] = {z8, z8};
#pragma unroll
        for (int ksq = 0; ksq < 3; ++ksq) {
            // B fragment: column n = d (fixed per lane), K = q strided in LDS
            v8u bv;
            const int qh = ksq * 32 + hi * 8;
#pragma unroll
            for (int rr = 0; rr < 8; ++rr) {
                unsigned lo = u_sh[(qh + 2 * rr + ((rr >= 4) ? 8 : 0))     * UB + dcol];
                unsigned hh = u_sh[(qh + 2 * rr + 1 + ((rr >= 4) ? 8 : 0)) * UB + dcol];
                bv[rr] = lo | (hh << 16);
            }
            v16bf bfr = __builtin_bit_cast(v16bf, bv);
            acc2[0] = __builtin_amdgcn_wmma_f32_16x16x32_bf16(
                false, a2[ksq][0], false, bfr, (short)0, acc2[0], false, false);
            acc2[1] = __builtin_amdgcn_wmma_f32_16x16x32_bf16(
                false, a2[ksq][1], false, bfr, (short)0, acc2[1], false, false);
        }
        const float hav = ha_sh[dcol];
#pragma unroll
        for (int mt = 0; mt < 2; ++mt) {
#pragma unroll
            for (int r = 0; r < 8; ++r) {
                const int m = mbase + mt * 16 + r + 8 * hi;
                const float ua = acc2[mt][r];
                const float hv = hB[(size_t)m * D_ + dcol];
                float* grow = gB + (size_t)m * G_;
                grow[dcol]            = hv;
                grow[D_ + dcol]       = ua;
                grow[2 * D_ + dcol]   = hv * ua;
                grow[3 * D_ + dcol]   = hv * hav;
            }
        }
    }
}

extern "C" void kernel_launch(void* const* d_in, const int* in_sizes, int n_in,
                              void* d_out, int out_size, void* d_ws, size_t ws_size,
                              hipStream_t stream) {
    (void)in_sizes; (void)n_in; (void)d_ws; (void)ws_size; (void)out_size;
    const float* h  = (const float*)d_in[0];
    const float* u  = (const float*)d_in[1];
    const float* hm = (const float*)d_in[2];
    const float* um = (const float*)d_in[3];
    // d_in[4] = is_train (unused: reference applies no dropout path here)
    const float* w  = (const float*)d_in[5];
    const float* bb = (const float*)d_in[6];
    float* g = (float*)d_out;

    attn_bidaf_gfx1250<<<dim3(B_ * S_), dim3(256), LDS_TOTAL, stream>>>(
        h, u, hm, um, w, bb, g);
}
